// CausalSelfAttention_84954453115585
// MI455X (gfx1250) — matrix operand order repaired and checked by reading, not compiled, not run on hardware
//
#include <hip/hip_runtime.h>

// ---------------------------------------------------------------------------
// CausalSelfAttention forward for MI455X (gfx1250), bf16 WMMA pipeline.
// B=4, T=2048, C=768, NH=12, HD=64.
// ---------------------------------------------------------------------------

#define BBATCH 4
#define TSEQ   2048
#define CDIM   768
#define C3     2304
#define NHEADS 12
#define HDIM   64
#define MROWS  (BBATCH * TSEQ)   // 8192
#define BH     (BBATCH * NHEADS) // 48

typedef __attribute__((ext_vector_type(16))) __bf16 v16bf;
typedef __attribute__((ext_vector_type(8)))  float  v8f;

union FragU { v16bf v; uint4 q[2]; };

static __device__ __forceinline__ unsigned short f2bf(float f) {
    unsigned int u = __float_as_uint(f);
    unsigned int r = ((u >> 16) & 1u) + 0x7FFFu; // round-to-nearest-even
    return (unsigned short)((u + r) >> 16);
}

// A-fragment (16x32 bf16): lane (l16,h) holds row M=l16, K runs
// [k0 + h*8, +8) and [k0 + 16 + h*8, +8)   (ISA 7.12.2, 16-bit A 16x32)
static __device__ __forceinline__ v16bf load_a_frag(const unsigned short* rowp, int k0, int h) {
    FragU f;
    f.q[0] = *(const uint4*)(rowp + k0 + h * 8);
    f.q[1] = *(const uint4*)(rowp + k0 + 16 + h * 8);
    return f.v;
}

// B-fragment (32x16 bf16): lane (l16,h) holds col N=l16, K runs
// [k0 + h*8, +8) and [k0 + 16 + h*8, +8). Memory layout must be [N][K] row-major.
static __device__ __forceinline__ v16bf load_b_frag(const unsigned short* colp, int k0, int h) {
    FragU f;
    f.q[0] = *(const uint4*)(colp + k0 + h * 8);
    f.q[1] = *(const uint4*)(colp + k0 + 16 + h * 8);
    return f.v;
}

static __device__ __forceinline__ v8f zero8() {
    v8f z;
#pragma unroll
    for (int i = 0; i < 8; ++i) z[i] = 0.0f;
    return z;
}

#define WMMA_BF16(a, b, c) \
    __builtin_amdgcn_wmma_f32_16x16x32_bf16(false, (a), false, (b), (short)0, (c), false, false)

// ---------------------------------------------------------------------------
// Conversion kernels
// ---------------------------------------------------------------------------
__global__ void cvt_bf16_kernel(const float* __restrict__ in, unsigned short* __restrict__ out, int n) {
    int i = blockIdx.x * blockDim.x + threadIdx.x;
    if (i < n) out[i] = f2bf(in[i]);
}

// W[K][N] (fp32) -> WT[N][K] (bf16)
__global__ void cvt_transpose_kernel(const float* __restrict__ W, unsigned short* __restrict__ WT,
                                     int K, int N) {
    int i = blockIdx.x * blockDim.x + threadIdx.x;
    if (i < K * N) {
        int n = i / K;
        int k = i - n * K;
        WT[i] = f2bf(W[(size_t)k * N + n]);
    }
}

// ---------------------------------------------------------------------------
// QKV GEMM: qkv = xb @ W_attn + b_attn,  M=8192 N=2304 K=768
// 64x64 tile per wave: 16 WMMAs per 16 b128 fragment loads per K-step.
// Scatter outputs into Q[BH][T][64], K[BH][T][64], Vt[BH][64][T]  (bf16)
// ---------------------------------------------------------------------------
__global__ void __launch_bounds__(256) qkv_gemm_kernel(
    const unsigned short* __restrict__ xb,   // [8192][768]
    const unsigned short* __restrict__ WaT,  // [2304][768]
    const float* __restrict__ b_attn,        // [2304]
    unsigned short* __restrict__ Qb,
    unsigned short* __restrict__ Kb,
    unsigned short* __restrict__ Vt) {
    const int lane = threadIdx.x & 31;
    const int l16 = lane & 15;
    const int h = lane >> 4;
    const int wid = blockIdx.x * 8 + (threadIdx.x >> 5);
    const int tm = wid / (C3 / 64);
    const int tn = wid % (C3 / 64);
    const int M0 = tm * 64, N0 = tn * 64;

    const unsigned short* arow[4];
    const unsigned short* bcol[4];
#pragma unroll
    for (int i = 0; i < 4; ++i) {
        arow[i] = xb + (size_t)(M0 + i * 16 + l16) * CDIM;
        bcol[i] = WaT + (size_t)(N0 + i * 16 + l16) * CDIM;
    }

    v8f acc[4][4];
#pragma unroll
    for (int mt = 0; mt < 4; ++mt)
#pragma unroll
        for (int nt = 0; nt < 4; ++nt) acc[mt][nt] = zero8();

#pragma unroll 2
    for (int k0 = 0; k0 < CDIM; k0 += 32) {
        v16bf a[4], b[4];
#pragma unroll
        for (int i = 0; i < 4; ++i) a[i] = load_a_frag(arow[i], k0, h);
#pragma unroll
        for (int i = 0; i < 4; ++i) b[i] = load_b_frag(bcol[i], k0, h);
#pragma unroll
        for (int mt = 0; mt < 4; ++mt)
#pragma unroll
            for (int nt = 0; nt < 4; ++nt)
                acc[mt][nt] = WMMA_BF16(a[mt], b[nt], acc[mt][nt]);
    }

#pragma unroll
    for (int nt = 0; nt < 4; ++nt) {
        const int j = N0 + nt * 16 + l16;   // column in [0, 2304)
        const float bias = b_attn[j];
        const int which = j / CDIM;         // 0=Q 1=K 2=V (uniform across tile)
        const int c = j - which * CDIM;
        const int head = c >> 6;
        const int d = c & 63;
#pragma unroll
        for (int mt = 0; mt < 4; ++mt) {
#pragma unroll
            for (int r = 0; r < 8; ++r) {
                const int t = M0 + mt * 16 + r + 8 * h; // global row [0, 8192)
                const int bb = t >> 11;
                const int tt = t & 2047;
                const int bh = bb * NHEADS + head;
                const unsigned short val = f2bf(acc[mt][nt][r] + bias);
                if (which == 0)
                    Qb[((size_t)bh * TSEQ + tt) * HDIM + d] = val;
                else if (which == 1)
                    Kb[((size_t)bh * TSEQ + tt) * HDIM + d] = val;
                else
                    Vt[((size_t)bh * HDIM + d) * TSEQ + tt] = val;
            }
        }
    }
}

// ---------------------------------------------------------------------------
// Flash attention: one wave per 16-row q tile. grid = BH * (T/16) = 6144
// ---------------------------------------------------------------------------
__global__ void __launch_bounds__(32) attn_kernel(
    const unsigned short* __restrict__ Qb,   // [BH][T][64]
    const unsigned short* __restrict__ Kb,   // [BH][T][64]
    const unsigned short* __restrict__ Vt,   // [BH][64][T]
    unsigned short* __restrict__ Yb) {       // [8192][768] bf16
    __shared__ unsigned short Plds[16 * 32];

    const int lane = threadIdx.x & 31;
    const int l16 = lane & 15;
    const int h = lane >> 4;
    const int bh = blockIdx.x >> 7;  // [0, 48)
    const int qt = blockIdx.x & 127;
    const int bb = bh / NHEADS;
    const int hh = bh - bb * NHEADS;

    // Q A-fragments for this tile (rows qt*16 .. qt*16+15, d = 0..63)
    const unsigned short* qrow = Qb + ((size_t)bh * TSEQ + qt * 16 + l16) * HDIM;
    const v16bf qa0 = load_a_frag(qrow, 0, h);
    const v16bf qa1 = load_a_frag(qrow, 32, h);

    float m[8], lsum[8];
#pragma unroll
    for (int r = 0; r < 8; ++r) { m[r] = -1e30f; lsum[r] = 0.0f; }
    v8f o[4];
#pragma unroll
    for (int dt = 0; dt < 4; ++dt) o[dt] = zero8();

    const int nChunks = (qt * 16 + 47) >> 5; // causal: only k <= q_max
    for (int c = 0; c < nChunks; ++c) {
        const int kbase = c * 32;
        v8f s0 = zero8(), s1 = zero8();
        {
            const unsigned short* krow0 = Kb + ((size_t)bh * TSEQ + kbase + l16) * HDIM;
            const unsigned short* krow1 = Kb + ((size_t)bh * TSEQ + kbase + 16 + l16) * HDIM;
            v16bf kb;
            kb = load_b_frag(krow0, 0, h);  s0 = WMMA_BF16(qa0, kb, s0);
            kb = load_b_frag(krow1, 0, h);  s1 = WMMA_BF16(qa0, kb, s1);
            kb = load_b_frag(krow0, 32, h); s0 = WMMA_BF16(qa1, kb, s0);
            kb = load_b_frag(krow1, 32, h); s1 = WMMA_BF16(qa1, kb, s1);
        }

        // scale + causal mask + online softmax (row M=r+8h lives on lanes of half h)
        float p0[8], p1[8], alpha[8];
#pragma unroll
        for (int r = 0; r < 8; ++r) {
            const int qg = qt * 16 + r + 8 * h;
            const int kg0 = kbase + l16;
            const int kg1 = kbase + 16 + l16;
            const float v0 = (kg0 <= qg) ? s0[r] * 0.125f : -1e30f;
            const float v1 = (kg1 <= qg) ? s1[r] * 0.125f : -1e30f;
            float rm = fmaxf(v0, v1);
            rm = fmaxf(rm, __shfl_xor(rm, 1));
            rm = fmaxf(rm, __shfl_xor(rm, 2));
            rm = fmaxf(rm, __shfl_xor(rm, 4));
            rm = fmaxf(rm, __shfl_xor(rm, 8));
            const float mnew = fmaxf(m[r], rm);
            alpha[r] = __expf(m[r] - mnew);
            m[r] = mnew;
            p0[r] = __expf(v0 - mnew);
            p1[r] = __expf(v1 - mnew);
            float ps = p0[r] + p1[r];
            ps += __shfl_xor(ps, 1);
            ps += __shfl_xor(ps, 2);
            ps += __shfl_xor(ps, 4);
            ps += __shfl_xor(ps, 8);
            lsum[r] = lsum[r] * alpha[r] + ps;
        }
#pragma unroll
        for (int dt = 0; dt < 4; ++dt)
#pragma unroll
            for (int r = 0; r < 8; ++r) o[dt][r] *= alpha[r];

        // P: D-layout -> LDS row-major [16][32] -> A-fragment
#pragma unroll
        for (int r = 0; r < 8; ++r) {
            const int row = r + 8 * h;
            Plds[row * 32 + l16] = f2bf(p0[r]);
            Plds[row * 32 + 16 + l16] = f2bf(p1[r]);
        }
        __syncthreads();
        v16bf pa;
        {
            FragU f;
            f.q[0] = *(const uint4*)(&Plds[l16 * 32 + h * 8]);
            f.q[1] = *(const uint4*)(&Plds[l16 * 32 + 16 + h * 8]);
            pa = f.v;
        }
        __syncthreads();

        // O += P @ V   (V stored transposed [d][t] => native B layout)
#pragma unroll
        for (int dt = 0; dt < 4; ++dt) {
            const unsigned short* vcol =
                Vt + ((size_t)bh * HDIM + dt * 16 + l16) * TSEQ + kbase;
            const v16bf vb = load_b_frag(vcol, 0, h);
            o[dt] = WMMA_BF16(pa, vb, o[dt]);
        }
    }

    // normalize and write y[b][t][hh*64 + d] as bf16
#pragma unroll
    for (int r = 0; r < 8; ++r) {
        const float inv = 1.0f / lsum[r];
        const int t = qt * 16 + r + 8 * h;
        const size_t base = ((size_t)bb * TSEQ + t) * CDIM + hh * HDIM;
        Yb[base + 0 * 16 + l16] = f2bf(o[0][r] * inv);
        Yb[base + 1 * 16 + l16] = f2bf(o[1][r] * inv);
        Yb[base + 2 * 16 + l16] = f2bf(o[2][r] * inv);
        Yb[base + 3 * 16 + l16] = f2bf(o[3][r] * inv);
    }
}

// ---------------------------------------------------------------------------
// Proj GEMM: out = y @ W_proj + b_proj, M=8192 N=768 K=768, fp32 out
// 32x64 tile per wave.
// ---------------------------------------------------------------------------
__global__ void __launch_bounds__(256) proj_gemm_kernel(
    const unsigned short* __restrict__ yb,   // [8192][768]
    const unsigned short* __restrict__ WpT,  // [768][768]
    const float* __restrict__ b_proj,        // [768]
    float* __restrict__ out) {               // [8192][768]
    const int lane = threadIdx.x & 31;
    const int l16 = lane & 15;
    const int h = lane >> 4;
    const int wid = blockIdx.x * 8 + (threadIdx.x >> 5);
    const int tm = wid / (CDIM / 64);
    const int tn = wid % (CDIM / 64);
    const int M0 = tm * 32, N0 = tn * 64;

    const unsigned short* arow[2];
    const unsigned short* bcol[4];
#pragma unroll
    for (int i = 0; i < 2; ++i) arow[i] = yb + (size_t)(M0 + i * 16 + l16) * CDIM;
#pragma unroll
    for (int i = 0; i < 4; ++i) bcol[i] = WpT + (size_t)(N0 + i * 16 + l16) * CDIM;

    v8f acc[2][4];
#pragma unroll
    for (int mt = 0; mt < 2; ++mt)
#pragma unroll
        for (int nt = 0; nt < 4; ++nt) acc[mt][nt] = zero8();

#pragma unroll 2
    for (int k0 = 0; k0 < CDIM; k0 += 32) {
        v16bf a[2], b[4];
#pragma unroll
        for (int i = 0; i < 2; ++i) a[i] = load_a_frag(arow[i], k0, h);
#pragma unroll
        for (int i = 0; i < 4; ++i) b[i] = load_b_frag(bcol[i], k0, h);
#pragma unroll
        for (int mt = 0; mt < 2; ++mt)
#pragma unroll
            for (int nt = 0; nt < 4; ++nt)
                acc[mt][nt] = WMMA_BF16(a[mt], b[nt], acc[mt][nt]);
    }

#pragma unroll
    for (int nt = 0; nt < 4; ++nt) {
        const int j = N0 + nt * 16 + l16;
        const float bias = b_proj[j];
#pragma unroll
        for (int mt = 0; mt < 2; ++mt) {
#pragma unroll
            for (int r = 0; r < 8; ++r) {
                const int t = M0 + mt * 16 + r + 8 * h;
                out[(size_t)t * CDIM + j] = acc[mt][nt][r] + bias;
            }
        }
    }
}

// ---------------------------------------------------------------------------
extern "C" void kernel_launch(void* const* d_in, const int* in_sizes, int n_in,
                              void* d_out, int out_size, void* d_ws, size_t ws_size,
                              hipStream_t stream) {
    (void)in_sizes; (void)n_in; (void)out_size; (void)ws_size;
    const float* x      = (const float*)d_in[0];
    const float* W_attn = (const float*)d_in[1];
    const float* b_attn = (const float*)d_in[2];
    const float* W_proj = (const float*)d_in[3];
    const float* b_proj = (const float*)d_in[4];
    float* out = (float*)d_out;

    // workspace layout (bf16 = unsigned short units)
    unsigned short* ws  = (unsigned short*)d_ws;
    unsigned short* XB  = ws;                              // 8192*768
    unsigned short* WAT = XB  + (size_t)MROWS * CDIM;      // 2304*768
    unsigned short* WPT = WAT + (size_t)C3 * CDIM;         // 768*768
    unsigned short* QB  = WPT + (size_t)CDIM * CDIM;       // 48*2048*64
    unsigned short* KB  = QB  + (size_t)BH * TSEQ * HDIM;
    unsigned short* VT  = KB  + (size_t)BH * TSEQ * HDIM;
    unsigned short* YB  = XB;  // reuse: x-bf16 is dead once QKV GEMM is done

    {
        const int n = MROWS * CDIM;
        cvt_bf16_kernel<<<(n + 255) / 256, 256, 0, stream>>>(x, XB, n);
    }
    {
        const int n = CDIM * C3;
        cvt_transpose_kernel<<<(n + 255) / 256, 256, 0, stream>>>(W_attn, WAT, CDIM, C3);
    }
    {
        const int n = CDIM * CDIM;
        cvt_transpose_kernel<<<(n + 255) / 256, 256, 0, stream>>>(W_proj, WPT, CDIM, CDIM);
    }
    // 8192/64 * 2304/64 = 4608 wave tiles / 8 waves per block
    qkv_gemm_kernel<<<(MROWS / 64) * (C3 / 64) / 8, 256, 0, stream>>>(XB, WAT, b_attn, QB, KB, VT);
    // 48 * 128 q-tiles, one wave each
    attn_kernel<<<BH * (TSEQ / 16), 32, 0, stream>>>(QB, KB, VT, YB);
    // 8192/32 * 768/64 = 3072 wave tiles / 8 waves per block
    proj_gemm_kernel<<<(MROWS / 32) * (CDIM / 64) / 8, 256, 0, stream>>>(YB, WPT, b_proj, out);
}
